// GPTLanguageModel_77455440216271
// MI455X (gfx1250) — matrix operand order repaired and checked by reading, not compiled, not run on hardware
//
#include <hip/hip_runtime.h>

// GPT-2-ish forward: B=4, T=1024, V=32000, D=1024, H=16, HS=64, FF=4096, L=8
#define Bq  4
#define Tt  1024
#define Vv  32000
#define Dd  1024
#define Hh  16
#define HSs 64
#define FFf 4096
#define Ll  8

typedef __attribute__((ext_vector_type(16))) __bf16          v16bf;
typedef __attribute__((ext_vector_type(16))) unsigned short  v16u;
typedef __attribute__((ext_vector_type(8)))  float           v8f;

union FragBF { v16u u; v16bf b; };

__device__ __forceinline__ unsigned short f2bf(float f) {
  unsigned int b = __float_as_uint(f);
  b += 0x7FFFu + ((b >> 16) & 1u);           // round-to-nearest-even
  return (unsigned short)(b >> 16);
}

// Issue one per-lane async 16B copy global -> LDS (CDNA5, ASYNCcnt-tracked).
__device__ __forceinline__ void async_b128(unsigned lds_off, const void* gp) {
  asm volatile("global_load_async_to_lds_b128 %0, %1, off"
               :: "v"(lds_off), "v"((unsigned long long)(size_t)gp)
               : "memory");
}
__device__ __forceinline__ void wait_async0() {
  asm volatile("s_wait_asynccnt 0x0" ::: "memory");
}

// Stage a 128x32 bf16 tile (row-major src, leading dim ld) into an LDS buffer
// with row stride 48 halves (96B; keeps all fragment reads 16B-aligned).
// 256 threads x 2 iters = 512 x b128 chunks = 8KB, no VGPR round trip.
__device__ __forceinline__ void stage_tile(unsigned lds_base,
                                           const unsigned short* src,
                                           long long ld, int tid) {
  #pragma unroll
  for (int it = 0; it < 2; ++it) {
    const int c   = tid + it * 256;
    const int row = c >> 2, col = (c & 3) * 8;
    async_b128(lds_base + (unsigned)(row * 48 + col) * 2u,
               src + (long long)row * ld + col);
  }
}

// ---------------------------------------------------------------------------
// bf16 WMMA GEMM: C[M,N] = act( A[M,K] * B^T + bias ) (+ residual)
// A row-major [M,K] (lda).  B stored [N,K] row-major (ldb) -- all weights are
// pre-transposed, K/V come naturally in this layout.
// Batched via grid.z: (b,h) = (z/nH, z%nH) strides for A/B/C.
// 128x128 block tile, BK=32, 8 waves each 32x64 (2x4 WMMA frags),
// double-buffered LDS filled by GLOBAL_LOAD_ASYNC_TO_LDS_B128.
// M, and N of every launch except PV, are tile-exact; OOB tile *reads* in the
// PV case land in adjacent workspace (harmless) and are masked at the store.
// ---------------------------------------------------------------------------
template<bool OUT_BF16, bool RELU, bool RES, bool OUT_VT>
__global__ __launch_bounds__(256) void gemm_wmma_kernel(
    const unsigned short* __restrict__ Ag, int lda, long long sAb, long long sAh,
    const unsigned short* __restrict__ Bg, int ldb, long long sBb, long long sBh,
    const float* __restrict__ bias,
    const float* __restrict__ resid,
    void* __restrict__ Cp, int ldc, long long sCb, long long sCh,
    int M, int N, int K, int nH)
{
  __shared__ __align__(16) unsigned short As[2][128 * 48];
  __shared__ __align__(16) unsigned short Bs[2][128 * 48];

  const int tid = threadIdx.x;
  const int bz  = blockIdx.z;
  const int bb  = bz / nH, hh = bz - bb * nH;
  const long long coff = (long long)bb * sCb + (long long)hh * sCh;

  const int m0 = blockIdx.y * 128;
  const int n0 = blockIdx.x * 128;
  const unsigned short* Arow = Ag + (long long)bb * sAb + (long long)hh * sAh
                                  + (long long)m0 * lda;
  const unsigned short* Brow = Bg + (long long)bb * sBb + (long long)hh * sBh
                                  + (long long)n0 * ldb;

  const unsigned aBase0 = (unsigned)(size_t)&As[0][0];
  const unsigned aBase1 = (unsigned)(size_t)&As[1][0];
  const unsigned bBase0 = (unsigned)(size_t)&Bs[0][0];
  const unsigned bBase1 = (unsigned)(size_t)&Bs[1][0];

  const int  wave  = tid >> 5;
  const int  lane  = tid & 31;
  const int  lr    = lane & 15;
  const bool hi    = lane >= 16;
  const int  waveM = (wave >> 1) * 32;   // 4 waves down M
  const int  waveN = (wave & 1) * 64;    // 2 waves across N

  v8f acc[2][4] = {};

  // prologue: fill buffer 0
  stage_tile(aBase0, Arow, lda, tid);
  stage_tile(bBase0, Brow, ldb, tid);
  wait_async0();
  __syncthreads();

  int cur = 0;
  for (int kb = 0; kb < K; kb += 32) {
    // async-prefetch next K tile into the other buffer while we compute
    if (kb + 32 < K) {
      stage_tile(cur ? aBase0 : aBase1, Arow + kb + 32, lda, tid);
      stage_tile(cur ? bBase0 : bBase1, Brow + kb + 32, ldb, tid);
    }

    const unsigned short* Ab = As[cur];
    const unsigned short* Bb = Bs[cur];

    // A 16x32 frag: lanes 0-15 hold K 0-15 of row lr; lanes 16-31 K 16-31
    FragBF af[2];
    #pragma unroll
    for (int i = 0; i < 2; ++i) {
      const unsigned short* p = &Ab[(waveM + i * 16 + lr) * 48 + (hi ? 16 : 0)];
      #pragma unroll
      for (int j = 0; j < 8; ++j) af[i].u[j] = p[j];
      #pragma unroll
      for (int j = 0; j < 8; ++j) af[i].u[8 + j] = p[8 + j];
    }
    // B 32x16 frag: lanes 0-15 hold K 0-15 of column lr; lanes 16-31 K 16-31
    FragBF bf[4];
    #pragma unroll
    for (int j = 0; j < 4; ++j) {
      const unsigned short* p = &Bb[(waveN + j * 16 + lr) * 48 + (hi ? 16 : 0)];
      #pragma unroll
      for (int t = 0; t < 16; ++t) bf[j].u[t] = p[t];
    }
    #pragma unroll
    for (int i = 0; i < 2; ++i)
      #pragma unroll
      for (int j = 0; j < 4; ++j)
        acc[i][j] = __builtin_amdgcn_wmma_f32_16x16x32_bf16(
            false, af[i].b, false, bf[j].b, (short)0, acc[i][j], false, false);

    wait_async0();       // next tile's copies have landed
    __syncthreads();     // everyone done reading cur, next buffer visible
    cur ^= 1;
  }

  // ---- epilogue: bias / relu / residual / store (C frag: VGPR r -> M=r | 8+r)
  #pragma unroll
  for (int j = 0; j < 4; ++j) {
    const int colg = n0 + waveN + j * 16 + lr;
    if (colg >= N) continue;
    const float bset = bias ? bias[colg] : 0.0f;
    #pragma unroll
    for (int i = 0; i < 2; ++i) {
      #pragma unroll
      for (int r = 0; r < 8; ++r) {
        const int rowg = m0 + waveM + i * 16 + (hi ? 8 + r : r);
        float v = acc[i][j][r] + bset;
        if (RELU) v = fmaxf(v, 0.0f);
        if (OUT_VT) {
          // write V transposed: vT[b, d=colg, t], rowg = b*T + t
          const long long vi = (long long)(rowg >> 10) * Dd * Tt
                             + (long long)colg * Tt + (rowg & (Tt - 1));
          ((unsigned short*)Cp)[vi] = f2bf(v);
        } else {
          const long long ci = coff + (long long)rowg * ldc + colg;
          if (RES) v += resid[ci];
          if (OUT_BF16) ((unsigned short*)Cp)[ci] = f2bf(v);
          else          ((float*)Cp)[ci]          = v;
        }
      }
    }
  }
}

// ---------------------------------------------------------------------------
// Embedding: x[b,t,:] = tok_emb[idx[b,t]] + pos_emb[t]
// ---------------------------------------------------------------------------
__global__ __launch_bounds__(256) void embed_kernel(
    const int* __restrict__ idx, const float* __restrict__ tok,
    const float* __restrict__ pos, float* __restrict__ x)
{
  const int bt = blockIdx.x;
  const int t  = bt & (Tt - 1);
  const long long tokoff = (long long)idx[bt] * Dd;
  for (int i = threadIdx.x; i < Dd; i += 256)
    x[(long long)bt * Dd + i] = tok[tokoff + i] + pos[(long long)t * Dd + i];
}

// ---------------------------------------------------------------------------
// LayerNorm over D=1024, one block per row, bf16 output (GEMM operand)
// ---------------------------------------------------------------------------
__global__ __launch_bounds__(256) void ln_kernel(
    const float* __restrict__ x, const float* __restrict__ g,
    const float* __restrict__ b, unsigned short* __restrict__ out)
{
  __shared__ float red[256];
  const int row = blockIdx.x;
  const int tid = threadIdx.x;
  const float* xr = x + (long long)row * Dd;
  float vals[4]; float s = 0.f;
  #pragma unroll
  for (int i = 0; i < 4; ++i) { vals[i] = xr[i * 256 + tid]; s += vals[i]; }
  red[tid] = s; __syncthreads();
  for (int off = 128; off > 0; off >>= 1) { if (tid < off) red[tid] += red[tid + off]; __syncthreads(); }
  const float mean = red[0] * (1.0f / Dd); __syncthreads();
  float s2 = 0.f;
  #pragma unroll
  for (int i = 0; i < 4; ++i) { const float d = vals[i] - mean; s2 += d * d; }
  red[tid] = s2; __syncthreads();
  for (int off = 128; off > 0; off >>= 1) { if (tid < off) red[tid] += red[tid + off]; __syncthreads(); }
  const float rstd = rsqrtf(red[0] * (1.0f / Dd) + 1e-5f);
  #pragma unroll
  for (int i = 0; i < 4; ++i) {
    const int c = i * 256 + tid;
    out[(long long)row * Dd + c] = f2bf((vals[i] - mean) * rstd * g[c] + b[c]);
  }
}

// ---------------------------------------------------------------------------
// Causal softmax over one score row of length T; scale=1/sqrt(HS) folded in.
// Masked entries written as exact bf16 zeros so the PV GEMM over full K works.
// ---------------------------------------------------------------------------
__global__ __launch_bounds__(256) void softmax_kernel(
    const float* __restrict__ scores, unsigned short* __restrict__ probs)
{
  __shared__ float red[256];
  const long long row = blockIdx.x;            // over B*H*T rows
  const int t   = (int)(row & (Tt - 1));
  const int tid = threadIdx.x;
  const float* sr = scores + row * Tt;
  const float scale = 0.125f;                  // HS^-0.5 = 1/8
  float vals[4]; float mx = -3.0e38f;
  #pragma unroll
  for (int i = 0; i < 4; ++i) {
    const int c = i * 256 + tid;
    vals[i] = (c <= t) ? sr[c] * scale : -3.0e38f;
    mx = fmaxf(mx, vals[i]);
  }
  red[tid] = mx; __syncthreads();
  for (int off = 128; off > 0; off >>= 1) { if (tid < off) red[tid] = fmaxf(red[tid], red[tid + off]); __syncthreads(); }
  const float rowmax = red[0]; __syncthreads();
  float sum = 0.f;
  #pragma unroll
  for (int i = 0; i < 4; ++i) {
    const int c = i * 256 + tid;
    vals[i] = (c <= t) ? __expf(vals[i] - rowmax) : 0.0f;
    sum += vals[i];
  }
  red[tid] = sum; __syncthreads();
  for (int off = 128; off > 0; off >>= 1) { if (tid < off) red[tid] += red[tid + off]; __syncthreads(); }
  const float inv = 1.0f / red[0];
  unsigned short* pr = probs + row * Tt;
  #pragma unroll
  for (int i = 0; i < 4; ++i) pr[i * 256 + tid] = f2bf(vals[i] * inv);
}

// ---------------------------------------------------------------------------
// Tiled transpose + f32->bf16: out[l][n][k] = bf16(in[l][k][n])
// Kd, Nd multiples of 32. Block (32,8), grid (Kd/32, Nd/32, L).
// ---------------------------------------------------------------------------
__global__ __launch_bounds__(256) void cvtT_kernel(
    const float* __restrict__ in, unsigned short* __restrict__ out,
    int Kd, int Nd)
{
  __shared__ float tile[32][33];
  const int l  = blockIdx.z;
  const int k0 = blockIdx.x * 32, n0 = blockIdx.y * 32;
  const float*    src = in  + (long long)l * Kd * Nd;
  unsigned short* dst = out + (long long)l * Nd * Kd;
  const int tx = threadIdx.x;
  for (int r = threadIdx.y; r < 32; r += 8)
    tile[r][tx] = src[(long long)(k0 + r) * Nd + n0 + tx];
  __syncthreads();
  for (int r = threadIdx.y; r < 32; r += 8)
    dst[(long long)(n0 + r) * Kd + k0 + tx] = f2bf(tile[tx][r]);
}

// ---------------------------------------------------------------------------
extern "C" void kernel_launch(void* const* d_in, const int* in_sizes, int n_in,
                              void* d_out, int out_size, void* d_ws, size_t ws_size,
                              hipStream_t stream)
{
  (void)in_sizes; (void)n_in; (void)out_size; (void)ws_size;
  const int*   idx  = (const int*)  d_in[0];
  const float* tok  = (const float*)d_in[1];
  const float* pos  = (const float*)d_in[2];
  const float* Wq   = (const float*)d_in[3];
  const float* Wk   = (const float*)d_in[4];
  const float* Wv   = (const float*)d_in[5];
  const float* Wp   = (const float*)d_in[6];
  const float* bp   = (const float*)d_in[7];
  const float* ln1g = (const float*)d_in[8];
  const float* ln1b = (const float*)d_in[9];
  const float* ln2g = (const float*)d_in[10];
  const float* ln2b = (const float*)d_in[11];
  const float* W1   = (const float*)d_in[12];
  const float* b1   = (const float*)d_in[13];
  const float* W2   = (const float*)d_in[14];
  const float* b2   = (const float*)d_in[15];
  const float* lnfg = (const float*)d_in[16];
  const float* lnfb = (const float*)d_in[17];
  const float* Wlm  = (const float*)d_in[18];
  const float* blm  = (const float*)d_in[19];

  char* wsp = (char*)d_ws;
  auto carve = [&](size_t bytes) -> void* {
    void* p = (void*)wsp;
    wsp += (bytes + 255) & ~(size_t)255;
    return p;
  };
  const long long MT = (long long)Bq * Tt;   // 4096 token rows

  float*          xbuf   = (float*)         carve(MT * Dd * 4);                       // residual stream f32
  unsigned short* xnbuf  = (unsigned short*)carve(MT * Dd * 2);                       // LN output bf16
  unsigned short* qbuf   = (unsigned short*)carve(MT * Dd * 2);                       // [B,T,H*HS]
  unsigned short* kbuf   = (unsigned short*)carve(MT * Dd * 2);                       // [B,T,H*HS]
  unsigned short* vtbuf  = (unsigned short*)carve(MT * Dd * 2);                       // V transposed [B,D,T]
  unsigned short* obuf   = (unsigned short*)carve(MT * Dd * 2);                       // attn out (head-concat)
  unsigned short* hbuf   = (unsigned short*)carve(MT * FFf * 2);                      // relu hidden bf16
  float*          scoreb = (float*)         carve((long long)Bq * Hh * Tt * Tt * 4);  // [B,H,T,T]
  unsigned short* probb  = (unsigned short*)carve((long long)Bq * Hh * Tt * Tt * 2);
  unsigned short* wqT    = (unsigned short*)carve((long long)Ll * Dd * Dd * 2);       // [L,N,K] bf16
  unsigned short* wkT    = (unsigned short*)carve((long long)Ll * Dd * Dd * 2);
  unsigned short* wvT    = (unsigned short*)carve((long long)Ll * Dd * Dd * 2);
  unsigned short* wpT    = (unsigned short*)carve((long long)Ll * Dd * Dd * 2);
  unsigned short* w1T    = (unsigned short*)carve((long long)Ll * Dd * FFf * 2);
  unsigned short* w2T    = (unsigned short*)carve((long long)Ll * FFf * Dd * 2);
  unsigned short* wlmT   = (unsigned short*)carve((long long)Dd * Vv * 2);

  // ---- weight transpose + bf16 conversion --------------------------------
  dim3 tb(32, 8);
  cvtT_kernel<<<dim3(Dd / 32, Dd / 32, Ll),  tb, 0, stream>>>(Wq, wqT, Dd, Dd);
  cvtT_kernel<<<dim3(Dd / 32, Dd / 32, Ll),  tb, 0, stream>>>(Wk, wkT, Dd, Dd);
  cvtT_kernel<<<dim3(Dd / 32, Dd / 32, Ll),  tb, 0, stream>>>(Wv, wvT, Dd, Dd);
  cvtT_kernel<<<dim3(Dd / 32, Dd / 32, Ll),  tb, 0, stream>>>(Wp, wpT, Dd, Dd);
  cvtT_kernel<<<dim3(Dd / 32, FFf / 32, Ll), tb, 0, stream>>>(W1, w1T, Dd, FFf);
  cvtT_kernel<<<dim3(FFf / 32, Dd / 32, Ll), tb, 0, stream>>>(W2, w2T, FFf, Dd);
  cvtT_kernel<<<dim3(Dd / 32, Vv / 32, 1),   tb, 0, stream>>>(Wlm, wlmT, Dd, Vv);

  embed_kernel<<<(int)MT, 256, 0, stream>>>(idx, tok, pos, xbuf);

  const long long sQb = (long long)Tt * Dd;       // per-batch stride in q/k/obuf
  const long long sSb = (long long)Hh * Tt * Tt;  // per-batch stride in scores/probs
  const long long sSh = (long long)Tt * Tt;       // per-head stride
  const long long sVb = (long long)Dd * Tt;       // per-batch stride in vT
  const long long sVh = (long long)HSs * Tt;      // per-head stride in vT

  for (int l = 0; l < Ll; ++l) {
    const unsigned short* wq_l = wqT + (long long)l * Dd * Dd;
    const unsigned short* wk_l = wkT + (long long)l * Dd * Dd;
    const unsigned short* wv_l = wvT + (long long)l * Dd * Dd;
    const unsigned short* wp_l = wpT + (long long)l * Dd * Dd;
    const unsigned short* w1_l = w1T + (long long)l * Dd * FFf;
    const unsigned short* w2_l = w2T + (long long)l * FFf * Dd;

    // LN1
    ln_kernel<<<(int)MT, 256, 0, stream>>>(xbuf, ln1g + (long long)l * Dd,
                                           ln1b + (long long)l * Dd, xnbuf);
    // Q/K/V projections: [4096,1024] x [1024,1024]^T
    dim3 gQKV(Dd / 128, (int)(MT / 128), 1);
    gemm_wmma_kernel<true, false, false, false><<<gQKV, 256, 0, stream>>>(
        xnbuf, Dd, 0, 0, wq_l, Dd, 0, 0, nullptr, nullptr,
        qbuf, Dd, 0, 0, (int)MT, Dd, Dd, 1);
    gemm_wmma_kernel<true, false, false, false><<<gQKV, 256, 0, stream>>>(
        xnbuf, Dd, 0, 0, wk_l, Dd, 0, 0, nullptr, nullptr,
        kbuf, Dd, 0, 0, (int)MT, Dd, Dd, 1);
    gemm_wmma_kernel<true, false, false, true><<<gQKV, 256, 0, stream>>>(
        xnbuf, Dd, 0, 0, wv_l, Dd, 0, 0, nullptr, nullptr,
        vtbuf, Dd, 0, 0, (int)MT, Dd, Dd, 1);      // -> vT[B,D,T]

    // scores = Q K^T per (b,h): [T,64] x [T,64]^T -> f32 [T,T]
    dim3 gS(Tt / 128, Tt / 128, Bq * Hh);
    gemm_wmma_kernel<false, false, false, false><<<gS, 256, 0, stream>>>(
        qbuf, Dd, sQb, HSs, kbuf, Dd, sQb, HSs, nullptr, nullptr,
        scoreb, Tt, sSb, sSh, Tt, Tt, HSs, Hh);

    softmax_kernel<<<Bq * Hh * Tt, 256, 0, stream>>>(scoreb, probb);

    // O = P V per (b,h): [T,T] x vT[64,T]^T -> bf16, head-concat [B,T,H*HS]
    dim3 gPV(1, Tt / 128, Bq * Hh);
    gemm_wmma_kernel<true, false, false, false><<<gPV, 256, 0, stream>>>(
        probb, Tt, sSb, sSh, vtbuf, Tt, sVb, sVh, nullptr, nullptr,
        obuf, Dd, sQb, HSs, Tt, HSs, Tt, Hh);

    // x += O Wp + bp  (fused residual)
    dim3 gPrj(Dd / 128, (int)(MT / 128), 1);
    gemm_wmma_kernel<false, false, true, false><<<gPrj, 256, 0, stream>>>(
        obuf, Dd, 0, 0, wp_l, Dd, 0, 0, bp + (long long)l * Dd, xbuf,
        xbuf, Dd, 0, 0, (int)MT, Dd, Dd, 1);

    // LN2
    ln_kernel<<<(int)MT, 256, 0, stream>>>(xbuf, ln2g + (long long)l * Dd,
                                           ln2b + (long long)l * Dd, xnbuf);
    // h = relu(xn W1 + b1) -> bf16
    dim3 gF1(FFf / 128, (int)(MT / 128), 1);
    gemm_wmma_kernel<true, true, false, false><<<gF1, 256, 0, stream>>>(
        xnbuf, Dd, 0, 0, w1_l, Dd, 0, 0, b1 + (long long)l * FFf, nullptr,
        hbuf, FFf, 0, 0, (int)MT, FFf, Dd, 1);
    // x += h W2 + b2  (fused residual)
    dim3 gF2(Dd / 128, (int)(MT / 128), 1);
    gemm_wmma_kernel<false, false, true, false><<<gF2, 256, 0, stream>>>(
        hbuf, FFf, 0, 0, w2_l, FFf, 0, 0, b2 + (long long)l * Dd, xbuf,
        xbuf, Dd, 0, 0, (int)MT, Dd, FFf, 1);
  }

  // final LN + LM head: [4096,1024] x [32000,1024]^T -> f32 logits
  ln_kernel<<<(int)MT, 256, 0, stream>>>(xbuf, lnfg, lnfb, xnbuf);
  dim3 gLM(Vv / 128, (int)(MT / 128), 1);
  gemm_wmma_kernel<false, false, false, false><<<gLM, 256, 0, stream>>>(
      xnbuf, Dd, 0, 0, wlmT, Dd, 0, 0, blm, nullptr,
      d_out, Vv, 0, 0, (int)MT, Vv, Dd, 1);
}
